// AttentionXL_55336358642567
// MI455X (gfx1250) — matrix operand order repaired and checked by reading, not compiled, not run on hardware
//
#include <hip/hip_runtime.h>
#include <hip/hip_bf16.h>

// ---------------------------------------------------------------------------
// TransformerXL attention on gfx1250 (MI455X), all matmuls via
// v_wmma_f32_16x16x32_f16 (f16 inputs, f32 accumulate).
//
// Pipeline (all on `stream`):
//   cvt:   full_input/inputs/pos_embedding f32 -> f16
//   trans: W_kv/W_q/W_pos/W_proj f32 (K x N) -> f16 (N x K)  ["Bt" layout]
//   G1: kv    = Xfull(8192x1024) @ Wkv   -> k[b,h,kk,d] f16 , valT[b,h,d,kk] f16
//   G2: q     = Xcur (4096x1024) @ Wq    -> qu[b,h,q,d], qv[b,h,q,d] f16 (+u,+v)
//   G3: r     = pos  (1024x1024) @ Wpos  -> r[h,kk,d] f16
//   G4: content logits[z,q,k]  = qu . k          (f32, *SCALE)     z=(b,h)
//   G5: position scatter-add via rel_shift bijection: dst = 1025*i'+k'+1-512
//   S:  masked softmax (vlen = q+513) -> attn f16
//   G6: attn_vec = attn @ valT  -> av[q,b,h,d] f16
//   G7: out = av @ Wproj + b_proj -> d_out f32
//
// CDNA5 data movement: double-buffered LDS staging via
// GLOBAL_LOAD_ASYNC_TO_LDS_B128 (ASYNCcnt / s_wait_asynccnt): tile t+1's
// fill overlaps tile t's WMMA work, one workgroup barrier per K-tile.
// XOR-swizzled LDS layout (stride 64 halfs) keeps fragment reads
// bank-conflict-free without padding. global_prefetch pre-pulls tile t+2.
// ---------------------------------------------------------------------------

typedef _Float16 h16;
typedef __attribute__((ext_vector_type(16))) _Float16 v16h;
typedef __attribute__((ext_vector_type(8)))  _Float16 v8h;
typedef __attribute__((ext_vector_type(8)))  float    v8f;
typedef int v4i __attribute__((__vector_size__(4 * sizeof(int))));

#define CURQ   512
#define FULLK  1024
#define BSZ    8
#define DIMM   1024
#define NH     16
#define HD     64
#define PREVK  512
#define SCL    0.125f   // 1/sqrt(64)

#define AS1 __attribute__((address_space(1)))
#define AS3 __attribute__((address_space(3)))

#if defined(__has_builtin)
#  if __has_builtin(__builtin_amdgcn_global_load_async_to_lds_b128) && \
      __has_builtin(__builtin_amdgcn_s_wait_asynccnt)
#    define HAVE_ASYNC_LDS 1
#  endif
#endif
#ifndef HAVE_ASYNC_LDS
#  define HAVE_ASYNC_LDS 0
#endif

#if HAVE_ASYNC_LDS
__device__ __forceinline__ void async_load_b128(const void* g, void* lds) {
    __builtin_amdgcn_global_load_async_to_lds_b128(
        (AS1 v4i*)(void*)g, (AS3 v4i*)lds, 0, 0);
}
#endif

// XOR-swizzled LDS offset (halfs). Row stride 64 halfs (128B); 16B chunk
// index is XORed with (row>>1)&7 so 16 consecutive rows hit distinct banks.
// kb must be a multiple of 8.
__device__ __forceinline__ int ldsoff(int r, int kb) {
    return r * 64 + (((kb >> 3) ^ ((r >> 1) & 7)) << 3);
}

// ---------------------------- elementwise kernels ---------------------------

__global__ void cvt_kernel(const float* __restrict__ src, h16* __restrict__ dst, int n) {
    int i = blockIdx.x * blockDim.x + threadIdx.x;
    int stride = gridDim.x * blockDim.x;
    for (; i < n; i += stride) dst[i] = (h16)src[i];
}

// W (K x N, row-major f32) -> Wt (N x K, row-major f16)
__global__ void trans_kernel(const float* __restrict__ W, h16* __restrict__ Wt, int K, int N) {
    int i = blockIdx.x * blockDim.x + threadIdx.x;
    int stride = gridDim.x * blockDim.x;
    int total = K * N;
    for (; i < total; i += stride) {
        int n = i / K;
        int k = i - n * K;
        Wt[i] = (h16)W[(size_t)k * N + n];
    }
}

// ------------------------------- WMMA GEMM ----------------------------------

struct EpiParams {
    const float* bias;
    const float* u;
    const float* v;
    h16*   out0;
    h16*   out1;
    float* outf;
    float  scale;
};

// C(MxN) = A(MxK) * Bt(NxK)^T ; A,Bt f16 row-major; 256 thr = 8 waves.
// BK = 64 (two 16x16x32 WMMA K-steps per LDS stage), double-buffered.
// Requires M%BM==0, N%BN==0, K%64==0 (true for every call below).
template<int BM, int BN, int WM, int WN, int EPI>
__launch_bounds__(256)
__global__ void gemm_wmma(const h16* __restrict__ A, const h16* __restrict__ Bt,
                          int M, int N, int K, int lda, int ldb,
                          long long strideA, long long strideB, int bmod,
                          EpiParams P)
{
    constexpr int BK    = 64;
    constexpr int WROWS = BM / WM;
    constexpr int WCOLS = BN / WN;
    static_assert(WROWS * WCOLS == 8, "need exactly 8 waves");
    constexpr int MF = WM / 16;
    constexpr int NF = WN / 16;

    __shared__ __align__(16) h16 As[2][BM * 64];
    __shared__ __align__(16) h16 Bs[2][BN * 64];

    const int tid  = threadIdx.x;
    const int lane = tid & 31;
    const int wid  = tid >> 5;
    const int wm   = wid % WROWS;
    const int wn   = wid / WROWS;
    const int wrow0 = wm * WM;
    const int wcol0 = wn * WN;

    const int n0 = blockIdx.x * BN;
    const int m0 = blockIdx.y * BM;
    const int z  = blockIdx.z;

    const h16* Az = A  + (long long)z * strideA;
    const h16* Bz = Bt + (long long)(bmod ? (z % bmod) : z) * strideB;

    v8f zero8 = {};
    v8f acc[MF][NF];
    for (int i = 0; i < MF; ++i)
        for (int j = 0; j < NF; ++j) acc[i][j] = zero8;

    const int laneRow = lane & 15;
    const int hihalf  = (lane & 16) ? 1 : 0;

    // stage one 64-wide K-tile of A and Bt into LDS buffer pb
    auto stage = [&](int kt, int pb) {
        for (int idx = tid; idx < BM * 8; idx += 256) {
            int r  = idx >> 3;
            int c8 = (idx & 7) * 8;
            const h16* g = Az + (long long)(m0 + r) * lda + kt + c8;
#if HAVE_ASYNC_LDS
            async_load_b128(g, &As[pb][ldsoff(r, c8)]);
#else
            *(v8h*)&As[pb][ldsoff(r, c8)] = *(const v8h*)g;
#endif
            if ((idx & 7) == 0 && kt + BK < K)
                __builtin_prefetch(g + BK, 0, 1);    // pre-pull next K-tile
        }
        for (int idx = tid; idx < BN * 8; idx += 256) {
            int r  = idx >> 3;
            int c8 = (idx & 7) * 8;
            const h16* g = Bz + (long long)(n0 + r) * ldb + kt + c8;
#if HAVE_ASYNC_LDS
            async_load_b128(g, &Bs[pb][ldsoff(r, c8)]);
#else
            *(v8h*)&Bs[pb][ldsoff(r, c8)] = *(const v8h*)g;
#endif
            if ((idx & 7) == 0 && kt + BK < K)
                __builtin_prefetch(g + BK, 0, 1);
        }
    };

    const int ntiles = K / BK;
    stage(0, 0);

    for (int t = 0; t < ntiles; ++t) {
#if HAVE_ASYNC_LDS
        __builtin_amdgcn_s_wait_asynccnt(0);   // own loads for buf t&1 done
#endif
        __syncthreads();                       // everyone's loads done
        if (t + 1 < ntiles)
            stage((t + 1) * BK, (t + 1) & 1);  // overlaps with compute below

        const h16* Ab = As[t & 1];
        const h16* Bb = Bs[t & 1];

        // ---- two 16x16x32 WMMA K-steps on the current buffer ----
        for (int ks = 0; ks < 2; ++ks) {
            v16h af[MF];
            v16h bf[NF];
            // A frag: lane holds row lane%16; K elems {b..b+7, 16+b..16+b+7}, b = hi?8:0
            for (int i = 0; i < MF; ++i) {
                int row = wrow0 + i * 16 + laneRow;
                int kb  = ks * 32 + hihalf * 8;
                ((v8h*)&af[i])[0] = *(const v8h*)&Ab[ldsoff(row, kb)];       // K kb..kb+7
                ((v8h*)&af[i])[1] = *(const v8h*)&Ab[ldsoff(row, kb + 16)];  // K kb+16..kb+23
            }
            // B frag: lane holds col lane%16; K elems {b..b+7, 16+b..16+b+7}, b = hi?8:0
            for (int j = 0; j < NF; ++j) {
                int row = wcol0 + j * 16 + laneRow;
                int kb  = ks * 32 + hihalf * 8;
                ((v8h*)&bf[j])[0] = *(const v8h*)&Bb[ldsoff(row, kb)];
                ((v8h*)&bf[j])[1] = *(const v8h*)&Bb[ldsoff(row, kb + 16)];
            }
            for (int i = 0; i < MF; ++i)
                for (int j = 0; j < NF; ++j)
                    acc[i][j] = __builtin_amdgcn_wmma_f32_16x16x32_f16(
                        false, af[i], false, bf[j], (short)0, acc[i][j], false, false);
        }
        // no trailing barrier: next iteration's wait+barrier protects buffer reuse
    }

    // ---- epilogue: C element (m,n): VGPR r -> M=r+(lane>=16?8:0), N=lane%16 ----
    const int ml = (lane & 16) ? 8 : 0;
    const int nl = lane & 15;
    for (int i = 0; i < MF; ++i) {
        for (int j = 0; j < NF; ++j) {
            for (int r = 0; r < 8; ++r) {
                int m = m0 + wrow0 + i * 16 + ml + r;
                int n = n0 + wcol0 + j * 16 + nl;
                float c = acc[i][j][r];
                if (EPI == 0) {                       // kv -> k[b,h,kk,d], valT[b,h,d,kk]
                    c += P.bias[n];
                    int kk = m >> 3, b = m & 7;
                    if (n < NH * HD) {
                        int h = n >> 6, d = n & 63;
                        P.out0[(((size_t)(b * NH + h) * FULLK + kk) << 6) + d] = (h16)c;
                    } else {
                        int nn = n - NH * HD;
                        int h = nn >> 6, d = nn & 63;
                        P.out1[((size_t)(b * NH + h) * HD + d) * FULLK + kk] = (h16)c;
                    }
                } else if (EPI == 1) {                // q -> qu[b,h,q,d], qv[b,h,q,d]
                    c += P.bias[n];
                    int qq = m >> 3, b = m & 7;
                    int h = n >> 6;
                    size_t o = (((size_t)(b * NH + h) * CURQ + qq) << 6) + (n & 63);
                    P.out0[o] = (h16)(c + P.u[n]);
                    P.out1[o] = (h16)(c + P.v[n]);
                } else if (EPI == 2) {                // r -> r[h,kk,d]
                    c += P.bias[n];
                    int h = n >> 6, d = n & 63;
                    P.out0[(((size_t)h * FULLK + m) << 6) + d] = (h16)c;
                } else if (EPI == 3) {                // content logits (scaled)
                    P.outf[(((size_t)z * CURQ + m) << 10) + n] = c * P.scale;
                } else if (EPI == 4) {                // position: rel_shift scatter-add
                    long long lin = 1025LL * m + n + 1 - PREVK;
                    if (lin >= 0 && lin < (long long)CURQ * FULLK) {
                        int ii = (int)(lin >> 10);
                        int jj = (int)(lin & 1023);
                        float* pd = &P.outf[(((size_t)z * CURQ + ii) << 10) + jj];
                        *pd += c * P.scale;          // bijective: no race
                    }
                } else if (EPI == 5) {                // attn_vec -> av[q,b,h,d]
                    int b = z >> 4, h = z & 15;
                    P.out0[((((size_t)m * BSZ + b) * NH + h) << 6) + n] = (h16)c;
                } else {                              // EPI == 6: final + b_proj
                    P.outf[(size_t)m * DIMM + n] = c + P.bias[n];
                }
            }
        }
    }
}

// ------------------------------- softmax ------------------------------------
// one 256-thread WG per (b,h,q) row; causal mask: valid length = q + 513
__global__ void softmax_kernel(const float* __restrict__ logits, h16* __restrict__ attn)
{
    __shared__ float sb[256];
    const int row  = blockIdx.x;             // row = z*CURQ + q
    const int qq   = row & (CURQ - 1);
    const int vlen = qq + PREVK + 1;         // <= 1024
    const float* x = logits + (size_t)row * FULLK;
    const int tid  = threadIdx.x;

    float lm = -3.0e38f;
    for (int j = tid; j < vlen; j += 256) lm = fmaxf(lm, x[j]);
    sb[tid] = lm; __syncthreads();
    for (int s = 128; s > 0; s >>= 1) {
        if (tid < s) sb[tid] = fmaxf(sb[tid], sb[tid + s]);
        __syncthreads();
    }
    float mx = sb[0]; __syncthreads();

    float ls = 0.f;
    for (int j = tid; j < vlen; j += 256) ls += __expf(x[j] - mx);
    sb[tid] = ls; __syncthreads();
    for (int s = 128; s > 0; s >>= 1) {
        if (tid < s) sb[tid] += sb[tid + s];
        __syncthreads();
    }
    float inv = 1.0f / sb[0];

    h16* o = attn + (size_t)row * FULLK;
    for (int j = tid; j < FULLK; j += 256)
        o[j] = (h16)((j < vlen) ? __expf(x[j] - mx) * inv : 0.f);
}

// ------------------------------- launcher -----------------------------------

extern "C" void kernel_launch(void* const* d_in, const int* in_sizes, int n_in,
                              void* d_out, int out_size, void* d_ws, size_t ws_size,
                              hipStream_t stream)
{
    (void)in_sizes; (void)n_in; (void)out_size; (void)ws_size;

    const float* inputs  = (const float*)d_in[0];
    const float* pos_emb = (const float*)d_in[1];
    const float* full_in = (const float*)d_in[2];
    const float* u       = (const float*)d_in[3];
    const float* v       = (const float*)d_in[4];
    /* d_in[5] = mask: recomputed analytically (vlen = q+513) */
    const float* W_kv    = (const float*)d_in[6];
    const float* b_kv    = (const float*)d_in[7];
    const float* W_q     = (const float*)d_in[8];
    const float* b_q     = (const float*)d_in[9];
    const float* W_pos   = (const float*)d_in[10];
    const float* b_pos   = (const float*)d_in[11];
    const float* W_proj  = (const float*)d_in[12];
    const float* b_proj  = (const float*)d_in[13];
    float* out = (float*)d_out;

    char* ws = (char*)d_ws;
    size_t off = 0;
    auto alloc = [&](size_t bytes) -> void* {
        void* p = ws + off;
        off += (bytes + 255) & ~(size_t)255;
        return p;
    };

    h16*  xf16   = (h16*)alloc(sizeof(h16) * (size_t)FULLK * BSZ * DIMM);
    h16*  xi16   = (h16*)alloc(sizeof(h16) * (size_t)CURQ * BSZ * DIMM);
    h16*  p16    = (h16*)alloc(sizeof(h16) * (size_t)FULLK * DIMM);
    h16*  wkvT   = (h16*)alloc(sizeof(h16) * (size_t)DIMM * 2 * NH * HD);
    h16*  wqT    = (h16*)alloc(sizeof(h16) * (size_t)DIMM * NH * HD);
    h16*  wposT  = (h16*)alloc(sizeof(h16) * (size_t)DIMM * NH * HD);
    h16*  wprojT = (h16*)alloc(sizeof(h16) * (size_t)NH * HD * DIMM);
    h16*  kbuf   = (h16*)alloc(sizeof(h16) * (size_t)BSZ * NH * FULLK * HD);
    h16*  valT   = (h16*)alloc(sizeof(h16) * (size_t)BSZ * NH * HD * FULLK);
    h16*  qu     = (h16*)alloc(sizeof(h16) * (size_t)BSZ * NH * CURQ * HD);
    h16*  qv     = (h16*)alloc(sizeof(h16) * (size_t)BSZ * NH * CURQ * HD);
    h16*  rbuf   = (h16*)alloc(sizeof(h16) * (size_t)NH * FULLK * HD);
    h16*  av16   = (h16*)alloc(sizeof(h16) * (size_t)CURQ * BSZ * NH * HD);
    float* logits = (float*)alloc(sizeof(float) * (size_t)BSZ * NH * CURQ * FULLK);
    h16*  attn   = (h16*)alloc(sizeof(h16) * (size_t)BSZ * NH * CURQ * FULLK);

    // f32 -> f16 conversions + weight transposes
    cvt_kernel<<<4096, 256, 0, stream>>>(full_in, xf16, FULLK * BSZ * DIMM);
    cvt_kernel<<<2048, 256, 0, stream>>>(inputs, xi16, CURQ * BSZ * DIMM);
    cvt_kernel<<<1024, 256, 0, stream>>>(pos_emb, p16, FULLK * DIMM);
    trans_kernel<<<2048, 256, 0, stream>>>(W_kv,   wkvT,   DIMM, 2 * NH * HD);
    trans_kernel<<<1024, 256, 0, stream>>>(W_q,    wqT,    DIMM, NH * HD);
    trans_kernel<<<1024, 256, 0, stream>>>(W_pos,  wposT,  DIMM, NH * HD);
    trans_kernel<<<1024, 256, 0, stream>>>(W_proj, wprojT, NH * HD, DIMM);

    EpiParams P{};

    // G1: kv projection -> k, valT
    P = {}; P.bias = b_kv; P.out0 = kbuf; P.out1 = valT;
    gemm_wmma<128, 128, 64, 32, 0><<<dim3(16, 64, 1), 256, 0, stream>>>(
        xf16, wkvT, FULLK * BSZ, 2 * NH * HD, DIMM, DIMM, DIMM, 0, 0, 0, P);

    // G2: q projection -> qu, qv
    P = {}; P.bias = b_q; P.u = u; P.v = v; P.out0 = qu; P.out1 = qv;
    gemm_wmma<128, 128, 64, 32, 1><<<dim3(8, 32, 1), 256, 0, stream>>>(
        xi16, wqT, CURQ * BSZ, NH * HD, DIMM, DIMM, DIMM, 0, 0, 0, P);

    // G3: position projection -> r
    P = {}; P.bias = b_pos; P.out0 = rbuf;
    gemm_wmma<128, 128, 64, 32, 2><<<dim3(8, 8, 1), 256, 0, stream>>>(
        p16, wposT, FULLK, NH * HD, DIMM, DIMM, DIMM, 0, 0, 0, P);

    // G4: content logits (z = b*H + h), scaled
    P = {}; P.outf = logits; P.scale = SCL;
    gemm_wmma<128, 128, 64, 32, 3><<<dim3(8, 4, BSZ * NH), 256, 0, stream>>>(
        qu, kbuf, CURQ, FULLK, HD, HD, HD,
        (long long)CURQ * HD, (long long)FULLK * HD, 0, P);

    // G5: position logits, rel_shift scatter-add (Bt indexed by h = z % 16)
    P = {}; P.outf = logits; P.scale = SCL;
    gemm_wmma<128, 128, 64, 32, 4><<<dim3(8, 4, BSZ * NH), 256, 0, stream>>>(
        qv, rbuf, CURQ, FULLK, HD, HD, HD,
        (long long)CURQ * HD, (long long)FULLK * HD, NH, P);

    // softmax with causal mask -> attn (f16)
    softmax_kernel<<<BSZ * NH * CURQ, 256, 0, stream>>>(logits, attn);

    // G6: attn @ val -> av[q,b,h,d]
    P = {}; P.out0 = av16;
    gemm_wmma<128, 64, 32, 32, 5><<<dim3(1, 4, BSZ * NH), 256, 0, stream>>>(
        attn, valT, CURQ, HD, FULLK, FULLK, FULLK,
        (long long)CURQ * FULLK, (long long)HD * FULLK, 0, P);

    // G7: output projection + b_proj -> d_out (f32)
    P = {}; P.bias = b_proj; P.outf = out;
    gemm_wmma<128, 128, 64, 32, 6><<<dim3(8, 32, 1), 256, 0, stream>>>(
        av16, wprojT, CURQ * BSZ, DIMM, NH * HD, NH * HD, NH * HD, 0, 0, 0, P);
}
